// DeformAttn_35356170780740
// MI455X (gfx1250) — matrix operand order repaired and checked by reading, not compiled, not run on hardware
//
#include <hip/hip_runtime.h>
#include <math.h>

#define B_  4
#define L_  8192
#define D_  512
#define H_  8
#define G_  4
#define BL_ (B_*L_)

typedef unsigned short u16;
typedef unsigned int   u32;
typedef u16    v8us  __attribute__((ext_vector_type(8)));
typedef u16    v16us __attribute__((ext_vector_type(16)));
typedef __bf16 v16bf __attribute__((ext_vector_type(16)));
typedef float  v8f   __attribute__((ext_vector_type(8)));
typedef int    v4i   __attribute__((ext_vector_type(4)));
typedef __attribute__((address_space(1))) v4i g_v4i;   // global
typedef __attribute__((address_space(3))) v4i s_v4i;   // LDS

#if __has_builtin(__builtin_amdgcn_global_load_async_to_lds_b128)
#define HAS_ASYNC 1
#else
#define HAS_ASYNC 0
#endif

// f32 -> bf16 via hardware convert (RNE), bits as u16
static __device__ inline u16 f2bf(float f) {
  return __builtin_bit_cast(u16, (__bf16)f);
}
static __device__ inline v16us cat16(v8us lo, v8us hi) {
  v16us r;
#pragma unroll
  for (int i = 0; i < 8; ++i) { r[i] = lo[i]; r[i + 8] = hi[i]; }
  return r;
}
static __device__ inline v8f wmma_bf16(v16us a, v16us b, v8f c) {
  return __builtin_amdgcn_wmma_f32_16x16x32_bf16(
      false, __builtin_bit_cast(v16bf, a),
      false, __builtin_bit_cast(v16bf, b),
      (short)0, c, false, false);
}

// 16-byte global -> LDS copy: async path (ASYNCcnt) when available
static __device__ inline void cp16B(u16* dst_lds, const u16* src) {
#if HAS_ASYNC
  __builtin_amdgcn_global_load_async_to_lds_b128(
      (g_v4i*)src, (s_v4i*)dst_lds, 0, 0);
#else
  *(v8us*)dst_lds = *(const v8us*)src;
#endif
}
// Wait until at most N of this wave's async copies remain outstanding.
template <int N>
static __device__ inline void wait_async_n() {
#if HAS_ASYNC
#if __has_builtin(__builtin_amdgcn_s_wait_asynccnt)
  __builtin_amdgcn_s_wait_asynccnt(N);
#else
  asm volatile("s_wait_asynccnt %0" :: "i"(N) : "memory");
#endif
#endif
}

// ---------------------------------------------------------------------------
// Elementwise f32 -> bf16 (4 elems/thread)
// ---------------------------------------------------------------------------
__global__ void __launch_bounds__(256)
cvt_kernel(const float* __restrict__ in, u16* __restrict__ out, int n)
{
  const int i = (blockIdx.x * 256 + threadIdx.x) * 4;
  if (i + 3 < n) {
#pragma unroll
    for (int j = 0; j < 4; ++j) out[i + j] = f2bf(in[i + j]);
  }
}

// ---------------------------------------------------------------------------
// GEMM: Y[r,o] = sum_d A[r,d]*W[o,d] + bias[o] (+ rel_bias[o*L + r%L])
// A,W pre-converted bf16. Block 128 = 4 waves (2x2), tile 64m x 128n, K=32.
// Double-buffered async staging: issue tile k+1, s_wait_asynccnt 6 (tile k's
// 6 in-order copies done), barrier, 8 WMMAs on tile k. HBM latency hidden.
// ---------------------------------------------------------------------------
template <bool RELB, bool WF32, bool WBF16>
__global__ void __launch_bounds__(128)
gemm_proj(const u16* __restrict__ Abf, const u16* __restrict__ Wbf,
          const float* __restrict__ bias, const float* __restrict__ relb,
          float* __restrict__ Yf, u16* __restrict__ Ybf)
{
  __shared__ __align__(16) u16 As[2][64 * 32];    // [buf][m][k]  2x4KB
  __shared__ __align__(16) u16 Bs[2][128 * 32];   // [buf][n][k]  2x8KB
  const int tid  = threadIdx.x;
  const int lane = tid & 31;
  const int wave = tid >> 5;
  const int r0 = blockIdx.x * 64;
  const int o0 = blockIdx.y * 128;
  const int wm = (wave >> 1) * 32;
  const int wn = (wave & 1) * 64;

  v8f acc[2][4];
#pragma unroll
  for (int mt = 0; mt < 2; ++mt)
#pragma unroll
    for (int nt = 0; nt < 4; ++nt)
#pragma unroll
      for (int i = 0; i < 8; ++i) acc[mt][nt][i] = 0.f;

  const int arow = tid >> 1, aseg = (tid & 1) * 16;

  auto stage = [&](int buf, int kk) {
    const u16* srcA = Abf + (size_t)(r0 + arow) * 512 + kk + aseg;
    const u16* srcB = Wbf + (size_t)(o0 + tid) * 512 + kk;
    cp16B(&As[buf][arow * 32 + aseg],     srcA);
    cp16B(&As[buf][arow * 32 + aseg + 8], srcA + 8);
#pragma unroll
    for (int j = 0; j < 4; ++j) cp16B(&Bs[buf][tid * 32 + j * 8], srcB + j * 8);
  };

  const int mrow  = lane & 15;
  const int khalf = lane >> 4;
  auto compute = [&](int buf) {
    v16us afr[2], bfr[4];
#pragma unroll
    for (int mt = 0; mt < 2; ++mt) {
      const int base = (wm + mt * 16 + mrow) * 32;
      afr[mt] = cat16(*(const v8us*)&As[buf][base + khalf * 8],
                      *(const v8us*)&As[buf][base + 16 + khalf * 8]);
    }
#pragma unroll
    for (int nt = 0; nt < 4; ++nt) {
      const int base = (wn + nt * 16 + mrow) * 32 + khalf * 8;
      bfr[nt] = cat16(*(const v8us*)&Bs[buf][base],
                      *(const v8us*)&Bs[buf][base + 16]);
    }
#pragma unroll
    for (int mt = 0; mt < 2; ++mt)
#pragma unroll
      for (int nt = 0; nt < 4; ++nt)
        acc[mt][nt] = wmma_bf16(afr[mt], bfr[nt], acc[mt][nt]);
  };

  stage(0, 0);
#pragma unroll 1
  for (int it = 0; it < 15; ++it) {
    stage((it + 1) & 1, (it + 1) * 32);   // prefetch next tile (async)
    wait_async_n<6>();                    // tile `it`'s copies complete
    __syncthreads();
    compute(it & 1);
    __syncthreads();                      // all reads done before overwrite
  }
  wait_async_n<0>();
  __syncthreads();
  compute(1);                             // it = 15

  const int nloc  = lane & 15;
  const int mhalf = (lane >> 4) * 8;
#pragma unroll
  for (int mt = 0; mt < 2; ++mt)
#pragma unroll
    for (int nt = 0; nt < 4; ++nt) {
      const int col = o0 + wn + nt * 16 + nloc;
      const float bcol = bias[col];
#pragma unroll
      for (int r = 0; r < 8; ++r) {
        const int row = r0 + wm + mt * 16 + mhalf + r;
        float v = acc[mt][nt][r] + bcol;
        if (RELB) v += relb[(size_t)col * L_ + (row & (L_ - 1))];
        if (WF32)  Yf [(size_t)row * 512 + col] = v;
        if (WBF16) Ybf[(size_t)row * 512 + col] = f2bf(v);
      }
    }
}

// ---------------------------------------------------------------------------
// Effective offset conv: Weff[ci,k] = sum_c w2[c]*W1[c,ci,k];
// weff[640]=beff=b2+dot(w2,b1); weff[641]=b2 (zero-pad case l<2)
// ---------------------------------------------------------------------------
__global__ void weff_kernel(const float* __restrict__ W1, const float* __restrict__ b1,
                            const float* __restrict__ w2, const float* __restrict__ b2,
                            float* __restrict__ weff)
{
  const int t = threadIdx.x;            // 0..639
  const int ci = t / 5, k = t % 5;
  float s = 0.f;
  for (int c = 0; c < 128; ++c) s += w2[c] * W1[(c * 128 + ci) * 5 + k];
  weff[t] = s;
  if (t == 0) {
    float be = b2[0];
    for (int c = 0; c < 128; ++c) be += w2[c] * b1[c];
    weff[640] = be;
    weff[641] = b2[0];
  }
}

// ---------------------------------------------------------------------------
// Offsets: one wave per (b,g,l) site; tanh(conv_eff(q)) -> sampling position.
// ---------------------------------------------------------------------------
__global__ void __launch_bounds__(256)
offset_kernel(const float* __restrict__ Q, const float* __restrict__ weff,
              float* __restrict__ pos_out)
{
  const int wave = threadIdx.x >> 5, lane = threadIdx.x & 31;
  const int site = blockIdx.x * 8 + wave;       // 0 .. 16*8192-1
  const int l  = site & (L_ - 1);
  const int bg = site >> 13;
  const int b = bg >> 2, g = bg & 3;
  float s;
  if (l < 2) {
    s = weff[641];                               // conv2 sees zero padding
  } else {
    float a = 0.f;
#pragma unroll
    for (int cc = 0; cc < 4; ++cc) {
      const int ci = lane + 32 * cc;
#pragma unroll
      for (int k = 0; k < 5; ++k) {
        const int t = l - 4 + k;
        if (t >= 0)
          a += weff[ci * 5 + k] * Q[(size_t)(b * L_ + t) * 512 + g * 128 + ci];
      }
    }
#pragma unroll
    for (int off = 16; off; off >>= 1) a += __shfl_xor(a, off, 32);
    s = a + weff[640];
  }
  if (lane == 0) {
    const float offs = tanhf(s) * 5.f;
    const float vg = (float)l + offs;
    const float gg = 2.f * vg / (float)(L_ + 3) - 1.f;  // n-1 = L+3
    pos_out[site] = ((gg + 1.f) * (float)L_ - 1.f) * 0.5f;
  }
}

// ---------------------------------------------------------------------------
// Bilinear gather with zero padding; writes packed bf16 pairs.
// One thread per channel pair (same group -> same position).
// ---------------------------------------------------------------------------
__global__ void __launch_bounds__(256)
gather_kernel(const float* __restrict__ X, const float* __restrict__ pos,
              u32* __restrict__ XSbf)
{
  const size_t pidx = (size_t)blockIdx.x * 256 + threadIdx.x;   // < BL*256
  const int d = (int)(pidx & 255) * 2;
  const size_t r = pidx >> 8;
  const int b = (int)(r >> 13), l = (int)(r & (L_ - 1));
  const int g = d >> 7;
  const float p = pos[(size_t)((b * 4 + g) << 13) + l];
  const float p0f = floorf(p);
  const float w1 = p - p0f;
  const int i0 = (int)p0f, i1 = i0 + 1;
  float a0 = 0.f, a1 = 0.f, c0 = 0.f, c1 = 0.f;
  if (i0 >= 0 && i0 < L_) {
    const float* s0 = X + ((size_t)b * L_ + i0) * 512 + d;
    a0 = s0[0]; a1 = s0[1];
  }
  if (i1 >= 0 && i1 < L_) {
    const float* s1 = X + ((size_t)b * L_ + i1) * 512 + d;
    c0 = s1[0]; c1 = s1[1];
  }
  const float r0 = a0 * (1.f - w1) + c0 * w1;
  const float r1 = a1 * (1.f - w1) + c1 * w1;
  XSbf[pidx] = (u32)f2bf(r0) | ((u32)f2bf(r1) << 16);
}

// ---------------------------------------------------------------------------
// Channel-attention scores: S = (qh @ kh^T) * D^-0.5 per (b,h); WMMA bf16.
// ---------------------------------------------------------------------------
__global__ void __launch_bounds__(128)
scores_kernel(const u16* __restrict__ Qbf, const u16* __restrict__ Kbf,
              float* __restrict__ S)
{
  __shared__ __align__(16) u16 Qs[64 * 32];    // [i][lk]
  __shared__ __align__(16) u16 Ks[64 * 32];    // [j][lk]
  const int bh = blockIdx.x;
  const int b = bh >> 3, h = bh & 7;
  const int tid = threadIdx.x, lane = tid & 31, wave = tid >> 5;

  v8f acc[4];
#pragma unroll
  for (int nt = 0; nt < 4; ++nt)
#pragma unroll
    for (int i = 0; i < 8; ++i) acc[nt][i] = 0.f;

  const int ch = tid & 63;
  const int lk0 = (tid >> 6) * 16;
  for (int l0 = 0; l0 < L_; l0 += 32) {
#pragma unroll
    for (int s = 0; s < 16; ++s) {
      const int lk = lk0 + s;
      const size_t off = (size_t)(b * L_ + l0 + lk) * 512 + h * 64 + ch;
      Qs[ch * 32 + lk] = Qbf[off];
      Ks[ch * 32 + lk] = Kbf[off];
    }
    __syncthreads();
    const int mrow = lane & 15, khalf = lane >> 4;
    const int abase = (wave * 16 + mrow) * 32;
    const v16us afr = cat16(*(const v8us*)&Qs[abase + khalf * 8],
                            *(const v8us*)&Qs[abase + 16 + khalf * 8]);
#pragma unroll
    for (int nt = 0; nt < 4; ++nt) {
      const int bbase = (nt * 16 + mrow) * 32 + khalf * 8;
      const v16us bfr = cat16(*(const v8us*)&Ks[bbase], *(const v8us*)&Ks[bbase + 16]);
      acc[nt] = wmma_bf16(afr, bfr, acc[nt]);
    }
    __syncthreads();
  }
  const float scale = 0.04419417382415922f;     // 512^-0.5
  const int nloc = lane & 15, mhalf = (lane >> 4) * 8;
#pragma unroll
  for (int nt = 0; nt < 4; ++nt)
#pragma unroll
    for (int r = 0; r < 8; ++r) {
      const int i = wave * 16 + mhalf + r;
      const int j = nt * 16 + nloc;
      S[(size_t)bh * 4096 + i * 64 + j] = acc[nt][r] * scale;
    }
}

// ---------------------------------------------------------------------------
// Row softmax over 64 (one wave per row; 2 elems/lane).
// ---------------------------------------------------------------------------
__global__ void __launch_bounds__(32)
softmax_kernel(float* __restrict__ S)
{
  const size_t row = blockIdx.x;
  const int lane = threadIdx.x;
  float a = S[row * 64 + lane];
  float c = S[row * 64 + 32 + lane];
  float m = fmaxf(a, c);
#pragma unroll
  for (int off = 16; off; off >>= 1) m = fmaxf(m, __shfl_xor(m, off, 32));
  const float e0 = __expf(a - m), e1 = __expf(c - m);
  float s = e0 + e1;
#pragma unroll
  for (int off = 16; off; off >>= 1) s += __shfl_xor(s, off, 32);
  const float inv = 1.f / s;
  S[row * 64 + lane] = e0 * inv;
  S[row * 64 + 32 + lane] = e1 * inv;
}

// ---------------------------------------------------------------------------
// out[b,l,h*64+i] = sum_j attn[bh,i,j]*v[b,l,h*64+j]; writes bf16 for outGEMM.
// ---------------------------------------------------------------------------
__global__ void __launch_bounds__(256)
av_kernel(const float* __restrict__ Attn, const float* __restrict__ V,
          u16* __restrict__ Obf)
{
  __shared__ float vrow[512];
  const int r = blockIdx.x;            // b*L + l
  const int b = r >> 13;
  const int t = threadIdx.x;
  vrow[t] = V[(size_t)r * 512 + t];
  vrow[t + 256] = V[(size_t)r * 512 + t + 256];
  __syncthreads();
#pragma unroll
  for (int rep = 0; rep < 2; ++rep) {
    const int d = t + rep * 256;
    const int h = d >> 6, i = d & 63;
    const float* arow = Attn + (size_t)(b * 8 + h) * 4096 + i * 64;
    float s = 0.f;
#pragma unroll 8
    for (int j = 0; j < 64; ++j) s += arow[j] * vrow[h * 64 + j];
    Obf[(size_t)r * 512 + d] = f2bf(s);
  }
}

// ---------------------------------------------------------------------------
extern "C" void kernel_launch(void* const* d_in, const int* in_sizes, int n_in,
                              void* d_out, int out_size, void* d_ws, size_t ws_size,
                              hipStream_t stream)
{
  const float* x     = (const float*)d_in[0];
  const float* Wq    = (const float*)d_in[1];
  const float* bq    = (const float*)d_in[2];
  const float* Wk    = (const float*)d_in[3];
  const float* bk    = (const float*)d_in[4];
  const float* Wv    = (const float*)d_in[5];
  const float* bv    = (const float*)d_in[6];
  const float* Woff1 = (const float*)d_in[7];
  const float* boff1 = (const float*)d_in[8];
  const float* Woff2 = (const float*)d_in[9];
  const float* boff2 = (const float*)d_in[10];
  const float* relb  = (const float*)d_in[11];
  const float* Wout  = (const float*)d_in[12];
  const float* bout  = (const float*)d_in[13];
  float* out = (float*)d_out;

  char* ws = (char*)d_ws;
  const size_t NF = (size_t)BL_ * D_;               // 16.7M elems
  const size_t F32 = NF * 4, BF = NF * 2, WB = (size_t)D_ * D_ * 2;
  float* q_f32  = (float*)(ws);                     // 64 MiB
  float* v_f32  = (float*)(ws + F32);               // 64 MiB
  u16*   x_bf   = (u16*)  (ws + 2 * F32);           // 32 MiB
  u16*   q_bf   = (u16*)  (ws + 2 * F32 + BF);      // 32 MiB
  u16*   k_bf   = (u16*)  (ws + 2 * F32 + 2 * BF);  // 32 MiB
  u16*   xs_bf  = (u16*)  (ws + 2 * F32 + 3 * BF);  // 32 MiB (reused as attnout)
  char*  tail   = ws + 2 * F32 + 4 * BF;
  u16*   Wq_bf  = (u16*)(tail);
  u16*   Wk_bf  = (u16*)(tail + WB);
  u16*   Wv_bf  = (u16*)(tail + 2 * WB);
  u16*   Wo_bf  = (u16*)(tail + 3 * WB);
  char*  tail2  = tail + 4 * WB;
  float* pos_ws = (float*)(tail2);                               // 16*8192 f32
  float* weff   = (float*)(tail2 + (size_t)(B_ * G_ * L_) * 4);  // 642 f32
  float* sc_ws  = (float*)(tail2 + (size_t)(B_ * G_ * L_) * 4 + 4096);

  const dim3 gemm_grid(BL_ / 64, D_ / 128);
  const int  NW = D_ * D_;                          // 262144

  cvt_kernel<<<(int)(NF / 1024), 256, 0, stream>>>(x, x_bf, (int)NF);
  cvt_kernel<<<NW / 1024, 256, 0, stream>>>(Wq, Wq_bf, NW);
  cvt_kernel<<<NW / 1024, 256, 0, stream>>>(Wk, Wk_bf, NW);
  cvt_kernel<<<NW / 1024, 256, 0, stream>>>(Wv, Wv_bf, NW);
  cvt_kernel<<<NW / 1024, 256, 0, stream>>>(Wout, Wo_bf, NW);
  weff_kernel<<<1, 640, 0, stream>>>(Woff1, boff1, Woff2, boff2, weff);

  gemm_proj<false, true, true><<<gemm_grid, 128, 0, stream>>>(
      x_bf, Wq_bf, bq, nullptr, q_f32, q_bf);
  offset_kernel<<<(B_ * G_ * L_) / 8, 256, 0, stream>>>(q_f32, weff, pos_ws);
  gather_kernel<<<(BL_ * 256) / 256, 256, 0, stream>>>(x, pos_ws, (u32*)xs_bf);
  gemm_proj<false, false, true><<<gemm_grid, 128, 0, stream>>>(
      xs_bf, Wk_bf, bk, nullptr, nullptr, k_bf);
  gemm_proj<true, true, false><<<gemm_grid, 128, 0, stream>>>(
      xs_bf, Wv_bf, bv, relb, v_f32, nullptr);
  scores_kernel<<<B_ * H_, 128, 0, stream>>>(q_bf, k_bf, sc_ws);
  softmax_kernel<<<B_ * H_ * 64, 32, 0, stream>>>(sc_ws);
  av_kernel<<<BL_, 256, 0, stream>>>(sc_ws, v_f32, xs_bf);   // xs reused
  gemm_proj<false, true, false><<<gemm_grid, 128, 0, stream>>>(
      xs_bf, Wo_bf, bout, nullptr, out, nullptr);
}
